// LSTM_2173253452080
// MI455X (gfx1250) — matrix operand order repaired and checked by reading, not compiled, not run on hardware
//
#include <hip/hip_runtime.h>

// Problem constants (match the reference).
#define T_ 512
#define B_ 64
#define H_ 1024
#define I_ 1024
#define NEXC_ 819          // int(0.8 * 1024)
#define ALPHA_ 0.2f
#define APITCH_ 1032       // LDS A-tile pitch (1024 + 8 elems = 16B pad -> 4-bank row skew)

typedef __attribute__((ext_vector_type(16))) __bf16        v16bf;
typedef __attribute__((ext_vector_type(8)))  float         v8f;
typedef __attribute__((ext_vector_type(8)))  unsigned int  v8u;
typedef __attribute__((ext_vector_type(4)))  unsigned int  v4u;
typedef __attribute__((ext_vector_type(4)))  float         vf4;

// ---------- bf16 helpers ----------
__device__ __forceinline__ unsigned bf_round(float f) {
  union { float f; unsigned u; } v; v.f = f;
  return (v.u + 0x7FFFu + ((v.u >> 16) & 1u)) >> 16;   // round-to-nearest-even
}
__device__ __forceinline__ unsigned pack_bf2(float lo, float hi) {
  return (bf_round(lo) & 0xFFFFu) | (bf_round(hi) << 16);
}
__device__ __forceinline__ float bf2f(unsigned short h) {
  union { unsigned u; float f; } v; v.u = ((unsigned)h) << 16;
  return v.f;
}

// ---------- WMMA ----------
__device__ __forceinline__ v8f wmma_bf16(v16bf a, v16bf b, v8f c) {
  return __builtin_amdgcn_wmma_f32_16x16x32_bf16(false, a, false, b, (short)0, c,
                                                 false, false);
}

// A fragment (16x32 bf16) from an f32 row, converting on the fly.
// lanes 0-15: M=lane, kb=0; lanes 16-31: same M set, kb=16.  kbase = k0 + kb.
// elems 0..15 <- row[kbase..+15].
__device__ __forceinline__ v16bf load_a_from_f32(const float* __restrict__ row, int kbase) {
  vf4 f0 = *(const vf4*)(row + kbase);
  vf4 f1 = *(const vf4*)(row + kbase + 4);
  vf4 f2 = *(const vf4*)(row + kbase + 8);
  vf4 f3 = *(const vf4*)(row + kbase + 12);
  v8u u;
  u[0] = pack_bf2(f0[0], f0[1]); u[1] = pack_bf2(f0[2], f0[3]);
  u[2] = pack_bf2(f1[0], f1[1]); u[3] = pack_bf2(f1[2], f1[3]);
  u[4] = pack_bf2(f2[0], f2[1]); u[5] = pack_bf2(f2[2], f2[3]);
  u[6] = pack_bf2(f3[0], f3[1]); u[7] = pack_bf2(f3[2], f3[3]);
  return __builtin_bit_cast(v16bf, u);
}

// A fragment straight from a bf16 row (global or LDS; two adjacent 16B runs).
__device__ __forceinline__ v16bf load_a_from_bf16(const unsigned short* __restrict__ row,
                                                  int kbase) {
  v4u lo = *(const v4u*)(row + kbase);
  v4u hi = *(const v4u*)(row + kbase + 8);
  v8u u;
  u[0] = lo[0]; u[1] = lo[1]; u[2] = lo[2]; u[3] = lo[3];
  u[4] = hi[0]; u[5] = hi[1]; u[6] = hi[2]; u[7] = hi[3];
  return __builtin_bit_cast(v16bf, u);
}

// B fragment (32x16, weights stored [N][K] row-major, i.e. A*W^T):
// lane holds column n = lane&15, K = k0 + (lane<16?0:16) .. +15 — one 32B run.
__device__ __forceinline__ v16bf load_b_bf16(const unsigned short* __restrict__ p) {
  v8u u = *(const v8u*)p;
  return __builtin_bit_cast(v16bf, u);
}

// ---------- kernel 0: rectify + Dale-sign weights into bf16 ----------
__global__ void prep_weights_kernel(const float* __restrict__ w_in,
                                    const float* __restrict__ w_rec,
                                    unsigned short* __restrict__ w_in_bf,
                                    unsigned short* __restrict__ w_rec_bf) {
  int idx = blockIdx.x * blockDim.x + threadIdx.x;
  if (idx >= H_ * I_) return;
  w_in_bf[idx] = (unsigned short)bf_round(fmaxf(w_in[idx], 0.f));
  int col = idx & (H_ - 1);                       // Dale sign per column j
  float sgn = (col < NEXC_) ? 1.f : -1.f;
  w_rec_bf[idx] = (unsigned short)bf_round(fmaxf(w_rec[idx], 0.f) * sgn);
}

// ---------- kernel 1: init state + barrier ----------
__global__ void init_kernel(float* __restrict__ sbuf, unsigned short* __restrict__ sbf,
                            unsigned* __restrict__ bar) {
  int idx = blockIdx.x * blockDim.x + threadIdx.x;
  if (idx < B_ * H_) { sbuf[idx] = 0.f; sbf[idx] = 0; }
  if (idx == 0) { bar[0] = 0u; bar[1] = 0u; }
}

// ---------- kernel 2: inp_upd = alpha * (emb[tokens] @ relu(w_in)^T), bf16 out ----------
// Wave computes a 16x64 strip (4 independent WMMA chains); K-loop software-pipelined.
__global__ void __launch_bounds__(256)
input_gemm_kernel(const int* __restrict__ tokens,
                  const float* __restrict__ emb,
                  const unsigned short* __restrict__ w_in_bf,
                  unsigned short* __restrict__ inp_upd_bf) {
  const int lane  = threadIdx.x & 31;
  const int wave  = threadIdx.x >> 5;
  const int wid   = blockIdx.x * 8 + wave;        // 0..32767
  const int mtile = wid >> 4;                     // 0..2047
  const int n0    = (wid & 15) * 64;              // 0..960
  const int l15   = lane & 15;

  const int mrow = mtile * 16 + l15;
  const float* arow = emb + (size_t)tokens[mrow] * I_;
  const int kbA = (lane < 16) ? 0 : 16;
  const int kbB = (lane < 16) ? 0 : 16;

  const unsigned short* b0 = w_in_bf + (size_t)(n0 +  0 + l15) * I_ + kbB;
  const unsigned short* b1 = w_in_bf + (size_t)(n0 + 16 + l15) * I_ + kbB;
  const unsigned short* b2 = w_in_bf + (size_t)(n0 + 32 + l15) * I_ + kbB;
  const unsigned short* b3 = w_in_bf + (size_t)(n0 + 48 + l15) * I_ + kbB;

  v8f acc0 = {}, acc1 = {}, acc2 = {}, acc3 = {};
  // Software pipeline: fragments for step i+1 load while WMMAs of step i run.
  v16bf a_c  = load_a_from_f32(arow, kbA);
  v16bf b0c = load_b_bf16(b0), b1c = load_b_bf16(b1);
  v16bf b2c = load_b_bf16(b2), b3c = load_b_bf16(b3);
#pragma unroll 2
  for (int k0 = 0; k0 < I_ - 32; k0 += 32) {
    const int kn = k0 + 32;
    v16bf a_n  = load_a_from_f32(arow, kn + kbA);
    v16bf b0n = load_b_bf16(b0 + kn), b1n = load_b_bf16(b1 + kn);
    v16bf b2n = load_b_bf16(b2 + kn), b3n = load_b_bf16(b3 + kn);
    acc0 = wmma_bf16(a_c, b0c, acc0);
    acc1 = wmma_bf16(a_c, b1c, acc1);
    acc2 = wmma_bf16(a_c, b2c, acc2);
    acc3 = wmma_bf16(a_c, b3c, acc3);
    a_c = a_n; b0c = b0n; b1c = b1n; b2c = b2n; b3c = b3n;
  }
  acc0 = wmma_bf16(a_c, b0c, acc0);
  acc1 = wmma_bf16(a_c, b1c, acc1);
  acc2 = wmma_bf16(a_c, b2c, acc2);
  acc3 = wmma_bf16(a_c, b3c, acc3);

  const int rbase = (lane < 16) ? 0 : 8;          // D rows: VGPR r -> M = r + rbase
#pragma unroll
  for (int r = 0; r < 8; ++r) {
    size_t m = (size_t)(mtile * 16 + rbase + r);
    unsigned short* orow = inp_upd_bf + m * H_ + n0 + l15;
    orow[0]  = (unsigned short)bf_round(ALPHA_ * acc0[r]);
    orow[16] = (unsigned short)bf_round(ALPHA_ * acc1[r]);
    orow[32] = (unsigned short)bf_round(ALPHA_ * acc2[r]);
    orow[48] = (unsigned short)bf_round(ALPHA_ * acc3[r]);
  }
}

// ---------- grid-wide barrier with release/acquire done by ALL threads ----------
__device__ __forceinline__ void grid_barrier(unsigned* bar, unsigned nblk) {
  __threadfence();                                  // release my stores to device scope
  __syncthreads();
  if (threadIdx.x == 0) {
    unsigned g = atomicAdd(&bar[1], 0u);            // read generation at L2
    if (atomicAdd(&bar[0], 1u) == nblk - 1u) {
      atomicExch(&bar[0], 0u);
      __threadfence();
      atomicAdd(&bar[1], 1u);                       // release generation
    } else {
      while (atomicAdd(&bar[1], 0u) == g) { __builtin_amdgcn_s_sleep(2); }
    }
  }
  __syncthreads();
  __threadfence();                                  // acquire: drop stale cached lines
}

// ---------- kernel 3: persistent 512-step leaky-RNN scan ----------
// 32 blocks x 8 waves (256 thr) = 256 waves = 256 (16x16) tiles of the [64,1024] state,
// spread over 32 WGPs so each SIMD only queues 64 WMMA issues per step.
// All 8 waves in a block share one 16-row M-tile of relu(s): staged in LDS each step.
// B fragments (w_rec, L2-resident) are software-pipelined 2 stages deep.
__global__ void __launch_bounds__(256)
recurrent_kernel(const unsigned short* __restrict__ inp_upd_bf,
                 const unsigned short* __restrict__ w_rec_bf,
                 float* __restrict__ sbuf,          // 2 * B_*H_ f32 (ping-pong)
                 unsigned short* __restrict__ sbf,  // 2 * B_*H_ bf16 relu-shadow
                 unsigned* __restrict__ bar) {
  __shared__ unsigned short aTile[16 * APITCH_];    // 33 KB, 4-bank row skew

  const int lane  = threadIdx.x & 31;
  const int wave  = threadIdx.x >> 5;
  const int wid   = blockIdx.x * 8 + wave;          // 0..255
  const int ntile = wid & 63;
  const int n0 = ntile * 16;
  const int m0 = (wid >> 6) * 16;                   // same for all waves in a block
  const int l15 = lane & 15;
  const int kbA = (lane < 16) ? 0 : 16;
  const int kbB = (lane < 16) ? 0 : 16;
  const int rbase = (lane < 16) ? 0 : 8;
  const unsigned short* bptr = w_rec_bf + (size_t)(n0 + l15) * H_ + kbB;
  const unsigned short* aLds = aTile + l15 * APITCH_;

  // LDS staging assignment: 256 threads x 64 elems = 16 x 1024 tile.
  const int srow = threadIdx.x >> 4;                // 0..15
  const int scol = (threadIdx.x & 15) * 64;         // 0..960
  unsigned short* sdst = aTile + srow * APITCH_ + scol;

  for (int t = 0; t < T_; ++t) {
    const float*          s_in   = sbuf + (size_t)(t & 1) * (B_ * H_);
    float*                s_out  = sbuf + (size_t)((t + 1) & 1) * (B_ * H_);
    const unsigned short* sb_in  = sbf  + (size_t)(t & 1) * (B_ * H_);
    unsigned short*       sb_out = sbf  + (size_t)((t + 1) & 1) * (B_ * H_);
    const unsigned short* iu     = inp_upd_bf + ((size_t)t * B_) * H_;

    // Stage this block's relu(s) M-tile into LDS (global b128 -> ds_store).
    {
      const unsigned short* ssrc = sb_in + (size_t)(m0 + srow) * H_ + scol;
#pragma unroll
      for (int j = 0; j < 8; ++j)
        *(v4u*)(sdst + j * 8) = *(const v4u*)(ssrc + j * 8);
    }
    __syncthreads();

    // Prefetch this step's inp_upd / s_in tile (HBM stream) behind the K-loop.
#pragma unroll
    for (int r = 0; r < 8; ++r) {
      size_t off = (size_t)(m0 + rbase + r) * H_ + n0 + l15;
      __builtin_prefetch(iu + off, 0, 0);
      __builtin_prefetch(s_in + off, 0, 0);
    }

    // K-loop: two interleaved accumulators (even/odd 32-chunks), 2-deep pipeline.
    v8f accA = {}, accB = {};
    v16bf aA0 = load_a_from_bf16(aLds,  0 + kbA), bA0 = load_b_bf16(bptr +  0);
    v16bf aB0 = load_a_from_bf16(aLds, 32 + kbA), bB0 = load_b_bf16(bptr + 32);
    v16bf aA1 = load_a_from_bf16(aLds, 64 + kbA), bA1 = load_b_bf16(bptr + 64);
    v16bf aB1 = load_a_from_bf16(aLds, 96 + kbA), bB1 = load_b_bf16(bptr + 96);
#pragma unroll 2
    for (int k0 = 128; k0 < H_; k0 += 64) {
      v16bf aA2 = load_a_from_bf16(aLds, k0 + kbA),      bA2 = load_b_bf16(bptr + k0);
      v16bf aB2 = load_a_from_bf16(aLds, k0 + 32 + kbA), bB2 = load_b_bf16(bptr + k0 + 32);
      accA = wmma_bf16(aA0, bA0, accA);
      accB = wmma_bf16(aB0, bB0, accB);
      aA0 = aA1; bA0 = bA1; aB0 = aB1; bB0 = bB1;
      aA1 = aA2; bA1 = bA2; aB1 = aB2; bB1 = bB2;
    }
    accA = wmma_bf16(aA0, bA0, accA);
    accB = wmma_bf16(aB0, bB0, accB);
    accA = wmma_bf16(aA1, bA1, accA);
    accB = wmma_bf16(aB1, bB1, accB);
    v8f acc = accA + accB;

#pragma unroll
    for (int r = 0; r < 8; ++r) {
      size_t off = (size_t)(m0 + rbase + r) * H_ + n0 + l15;
      float v = (1.f - ALPHA_) * s_in[off] + ALPHA_ * acc[r] + bf2f(iu[off]);
      s_out[off]  = v;
      sb_out[off] = (unsigned short)bf_round(fmaxf(v, 0.f));
    }
    grid_barrier(bar, gridDim.x);
  }
}

// ---------- kernel 4: logits = s_last @ lin_w^T + lin_b ----------
__global__ void head_kernel(const float* __restrict__ s_last,
                            const float* __restrict__ lin_w,
                            const float* __restrict__ lin_b,
                            float* __restrict__ out) {
  int tid = threadIdx.x;            // 128 threads: (b, k)
  int b = tid >> 1, k = tid & 1;
  const float* s = s_last + (size_t)b * H_;
  const float* w = lin_w + (size_t)k * H_;
  float acc = lin_b[k];
  for (int h = 0; h < H_; ++h) acc += s[h] * w[h];
  out[b * 2 + k] = acc;
}

// ---------- launch ----------
extern "C" void kernel_launch(void* const* d_in, const int* in_sizes, int n_in,
                              void* d_out, int out_size, void* d_ws, size_t ws_size,
                              hipStream_t stream) {
  (void)in_sizes; (void)n_in; (void)out_size; (void)ws_size;
  const int*   tokens = (const int*)d_in[0];
  const float* emb    = (const float*)d_in[1];
  const float* w_in   = (const float*)d_in[2];
  const float* w_rec  = (const float*)d_in[3];
  const float* lin_w  = (const float*)d_in[4];
  const float* lin_b  = (const float*)d_in[5];

  // Workspace layout (~69 MB total)
  char* ws = (char*)d_ws;
  size_t off = 0;
  unsigned short* inp_upd_bf = (unsigned short*)(ws + off); off += (size_t)T_ * B_ * H_ * 2;
  unsigned short* w_in_bf    = (unsigned short*)(ws + off); off += (size_t)H_ * I_ * 2;
  unsigned short* w_rec_bf   = (unsigned short*)(ws + off); off += (size_t)H_ * H_ * 2;
  float*          sbuf       = (float*)(ws + off);          off += (size_t)2 * B_ * H_ * 4;
  unsigned short* sbf        = (unsigned short*)(ws + off); off += (size_t)2 * B_ * H_ * 2;
  unsigned*       bar        = (unsigned*)(ws + off);

  prep_weights_kernel<<<(H_ * I_ + 255) / 256, 256, 0, stream>>>(w_in, w_rec, w_in_bf, w_rec_bf);
  init_kernel<<<(B_ * H_ + 255) / 256, 256, 0, stream>>>(sbuf, sbf, bar);
  input_gemm_kernel<<<4096, 256, 0, stream>>>(tokens, emb, w_in_bf, inp_upd_bf);
  recurrent_kernel<<<32, 256, 0, stream>>>(inp_upd_bf, w_rec_bf, sbuf, sbf, bar);
  // T_=512 is even -> final state lands in ping-pong buffer 0.
  head_kernel<<<1, 128, 0, stream>>>(sbuf, lin_w, lin_b, (float*)d_out);
}
